// MultiDetNNBackflowPure_62792421867573
// MI455X (gfx1250) — matrix operand order repaired and checked by reading, not compiled, not run on hardware
//
#include <hip/hip_runtime.h>
#include <hip/hip_bf16.h>
#include <math.h>

typedef __attribute__((ext_vector_type(16))) _Float16 v16h;
typedef __attribute__((ext_vector_type(8)))  float    v8f;

#define B_SZ   1024
#define NSITE  512
#define NFERM  64
#define KENS   4
#define WDIM   64
#define NMODE  64     // 8x8 retained modes
#define PI_F   3.14159265358979323846f

// gelu_tanh(x) = 0.5*x*(1+tanh(z)) = x * sigmoid(2z); branch-free:
// one v_exp_f32 + one v_rcp_f32, saturates correctly at both ends.
__device__ __forceinline__ float gelu_fast(float x){
    float u = 1.5957691216057308f * (x + 0.044715f*x*x*x);   // 2*sqrt(2/pi)*(x+c*x^3)
    float e = __expf(-u);                                    // v_exp_f32 (no branches)
    return x * __builtin_amdgcn_rcpf(1.0f + e);              // v_rcp_f32
}

// ---------------- k0: per-sample 8x8 DFT corner of n, plus occupied-site list ----------------
__global__ __launch_bounds__(64) void k0_modes_gather(const int* __restrict__ n,
                                                      float* __restrict__ nf,
                                                      int* __restrict__ R){
    int b = blockIdx.x;
    int t = threadIdx.x;                 // 64 threads: one (kx,ky) mode each
    __shared__ int ns[NSITE];
    for (int i = t; i < NSITE; i += 64) ns[i] = n[b*NSITE + i];
    __syncthreads();
    int kx = t >> 3, ky = t & 7;
    float re = 0.f, im = 0.f;
    for (int s = 0; s < NSITE; ++s){
        if (ns[s]){                       // uniform branch (shared memory), ~64 taken
            int h = s >> 5, w = s & 31;
            float ang = -2.0f*PI_F*((float)(kx*h)*(1.0f/16.0f) + (float)(ky*w)*(1.0f/32.0f));
            float sv, cv; __sincosf(ang, &sv, &cv);
            re += cv; im += sv;
        }
    }
    nf[(b*NMODE + t)*2 + 0] = re;
    nf[(b*NMODE + t)*2 + 1] = im;
    if (t == 0){
        int cnt = 0;
        for (int s = 0; s < NSITE && cnt < NFERM; ++s)
            if (ns[s]) R[b*NFERM + cnt++] = s;
    }
}

// ---------------- k1: pre-contract weights: Wc, wp, bp2, bterm ----------------
__global__ void k1_weights(const float* __restrict__ Wl,  const float* __restrict__ bl,
                           const float* __restrict__ Wre, const float* __restrict__ Wim,
                           const float* __restrict__ Wpt, const float* __restrict__ bpt,
                           float* __restrict__ Wc, float* __restrict__ bterm,
                           float* __restrict__ wp, float* __restrict__ bp2){
    int gid = blockIdx.x*blockDim.x + threadIdx.x;
    if (gid < KENS*WDIM*NMODE){
        int k = gid / (WDIM*NMODE);
        int o = (gid / NMODE) % WDIM;
        int m = gid % NMODE;
        float sre = 0.f, sim = 0.f;
        for (int i = 0; i < WDIM; ++i){
            float wl = Wl[k*WDIM + i];                    // C==1
            int idx = ((k*WDIM + i)*WDIM + o)*NMODE + m;  // (K,WDin,WDout,8,8)
            sre += wl*Wre[idx];
            sim += wl*Wim[idx];
        }
        Wc[gid*2+0] = sre; Wc[gid*2+1] = sim;
    }
    if (gid < KENS*WDIM){
        int k = gid / WDIM, o = gid % WDIM;
        float a = 0.f, b2 = 0.f, bre = 0.f, bim = 0.f;
        for (int i = 0; i < WDIM; ++i){
            float wl = Wl[k*WDIM + i], blv = bl[k*WDIM + i];
            float wptv = Wpt[(k*WDIM + i)*WDIM + o];
            a  += wl*wptv;
            b2 += blv*wptv;
            int idx0 = ((k*WDIM + i)*WDIM + o)*NMODE + 0; // mode (0,0)
            bre += blv*Wre[idx0];
            bim += blv*Wim[idx0];
        }
        wp[gid]  = a;
        bp2[gid] = b2 + bpt[gid];
        bterm[gid*2+0] = 512.0f*bre;   // FFT of constant bias -> (0,0) mode * H*W
        bterm[gid*2+1] = 512.0f*bim;
    }
}

// ---------------- kT: fixed irfft2 basis T[512][128] = [cos | sin], f16 ----------------
__global__ void kT_basis(_Float16* __restrict__ T){
    int gid = blockIdx.x*blockDim.x + threadIdx.x;   // 512*128
    if (gid >= NSITE*128) return;
    int hw = gid >> 7;
    int m  = gid & 127;
    int mm = m & 63;
    int kx = mm >> 3, ky = mm & 7;
    int h = hw >> 5, w = hw & 31;
    float ang = 2.0f*PI_F*((float)(kx*h)*(1.0f/16.0f) + (float)(ky*w)*(1.0f/32.0f));
    float sv, cv; __sincosf(ang, &sv, &cv);
    T[gid] = (_Float16)((m < 64) ? cv : sv);
}

// ---------------- k2: fused Spec = T @ Cc (WMMA), + pt, gelu, mean over hw -> feat ----------------
union ABPack { v16h v; uint4 u[2]; };

__global__ __launch_bounds__(256) void k2_fno_feat(const _Float16* __restrict__ Tg,
                 const float* __restrict__ Wc, const float* __restrict__ bterm,
                 const float* __restrict__ wp, const float* __restrict__ bp2,
                 const float* __restrict__ nf, const int* __restrict__ n,
                 float* __restrict__ feat){
    int bk = blockIdx.x;
    int b = bk >> 2, k = bk & 3;
    int tid  = threadIdx.x;
    int lane = tid & 31, wv = tid >> 5;
    __shared__ __align__(16) _Float16 CcT[WDIM][136];   // [o][m], padded rows (bank-spread)
    __shared__ int   ns[NSITE];
    __shared__ float nfs[NMODE*2];
    __shared__ float wps[WDIM], bp2s[WDIM], featsum[WDIM];

    for (int i = tid; i < NSITE;   i += 256) ns[i]  = n[b*NSITE + i];
    for (int i = tid; i < NMODE*2; i += 256) nfs[i] = nf[b*NMODE*2 + i];
    if (tid < WDIM){ wps[tid] = wp[k*WDIM+tid]; bp2s[tid] = bp2[k*WDIM+tid]; featsum[tid] = 0.f; }
    __syncthreads();

    // Cc[o][m<64]  =  alpha/512 * Re(nf_m * Wc[k,o,m] + d00*bterm)
    // Cc[o][64+m]  = -alpha/512 * Im(  ...same...  )
    for (int idx = tid; idx < WDIM*NMODE; idx += 256){
        int o = idx >> 6, m = idx & 63;
        float wre = Wc[((k*WDIM + o)*NMODE + m)*2 + 0];
        float wim = Wc[((k*WDIM + o)*NMODE + m)*2 + 1];
        float nre = nfs[m*2+0], nim = nfs[m*2+1];
        float cre = nre*wre - nim*wim;
        float cim = nre*wim + nim*wre;
        if (m == 0){ cre += bterm[(k*WDIM+o)*2+0]; cim += bterm[(k*WDIM+o)*2+1]; }
        int ky = m & 7;
        float s = ((ky == 0) ? 1.0f : 2.0f) * (1.0f/512.0f);
        CcT[o][m]      = (_Float16)( s*cre);
        CcT[o][64 + m] = (_Float16)(-s*cim);
    }
    __syncthreads();

    // GEMM: Spec(512x64) = T(512x128) @ Cc(128x64); M-tiles 32, N-tiles 4, K-steps 4 (16x16x32 f16)
    float lsum[4] = {0.f, 0.f, 0.f, 0.f};
    int halfsel = lane >> 4;        // 0/1 selects K-half per ISA operand layout
    int lrow    = lane & 15;
    for (int q = 0; q < 4; ++q){
        int mt = wv + 8*q;          // this wave's M-tile
        for (int nt = 0; nt < 4; ++nt){
            v8f acc = {0.f,0.f,0.f,0.f,0.f,0.f,0.f,0.f};
            #pragma unroll
            for (int ks = 0; ks < 4; ++ks){
                ABPack a, bb;
                // A 16x32 f16: lane holds K-chunks [halfsel*8..+8) and [16+halfsel*8..+8)
                const _Float16* arow = Tg + (mt*16 + lrow)*128 + ks*32 + halfsel*8;
                a.u[0] = *(const uint4*)(arow);
                a.u[1] = *(const uint4*)(arow + 16);
                // B 32x16 f16: lane holds col lrow, K-chunks [halfsel*8..+8) and [16+halfsel*8..+8)
                const _Float16* brow = &CcT[nt*16 + lrow][ks*32 + halfsel*8];
                bb.u[0] = *(const uint4*)(brow);
                bb.u[1] = *(const uint4*)(brow + 16);
                acc = __builtin_amdgcn_wmma_f32_16x16x32_f16(false, a.v, false, bb.v,
                                                             (short)0, acc, false, false);
            }
            // C layout: VGPR r, lanes 0-15 -> M=r, lanes 16-31 -> M=r+8; N = lane%16
            int oc  = nt*16 + lrow;
            float wpo = wps[oc], bpo = bp2s[oc];
            #pragma unroll
            for (int r = 0; r < 8; ++r){
                int hw = mt*16 + r + 8*halfsel;
                float x = acc[r] + bpo + (ns[hw] ? wpo : 0.0f);
                lsum[nt] += gelu_fast(x);
            }
        }
    }
    #pragma unroll
    for (int nt = 0; nt < 4; ++nt) atomicAdd(&featsum[nt*16 + lrow], lsum[nt]);
    __syncthreads();
    if (tid < WDIM) feat[(b*KENS + k)*WDIM + tid] = featsum[tid] * (1.0f/512.0f);
}

// ---------------- k3: gathered projection (A-build) + LU logdet, fully in LDS ----------------
__global__ __launch_bounds__(256) void k3_det(const float* __restrict__ feat, const int* __restrict__ R,
                       const float* __restrict__ Wproj, const float* __restrict__ bproj,
                       float* __restrict__ logabs, float* __restrict__ sgn){
    int bk = blockIdx.x; int b = bk >> 2, k = bk & 3;
    int tid = threadIdx.x;
    __shared__ float As[NFERM][NFERM + 1];
    __shared__ float fs[WDIM];
    __shared__ int   Rs[NFERM];
    __shared__ int   pivrow_s;
    __shared__ float pivinv_s;
    if (tid < WDIM)  fs[tid] = feat[(b*KENS + k)*WDIM + tid];
    if (tid < NFERM) Rs[tid] = R[b*NFERM + tid];
    __syncthreads();
    // A[i][j] = b_proj[k, R[i]*64+j] + sum_w feat[w] * W_proj[k, w, R[i]*64+j]
    for (int idx = tid; idx < NFERM*NFERM; idx += 256){
        int i = idx >> 6, j = idx & 63;
        int col = Rs[i]*NFERM + j;
        float acc = bproj[k*(NSITE*NFERM) + col];
        const float* wcol = Wproj + (size_t)k*WDIM*NSITE*NFERM + col;
        #pragma unroll 4
        for (int w = 0; w < WDIM; ++w) acc += fs[w]*wcol[(size_t)w*NSITE*NFERM];
        As[i][j] = acc;
    }
    __syncthreads();
    // LU with partial pivoting; det sign/logabs are pivot-invariant
    float la = 0.f; int sg = 1;
    for (int p = 0; p < NFERM; ++p){
        if (tid == 0){
            int best = p; float bv = fabsf(As[p][p]);
            for (int r = p + 1; r < NFERM; ++r){
                float v = fabsf(As[r][p]);
                if (v > bv){ bv = v; best = r; }
            }
            pivrow_s = best;
        }
        __syncthreads();
        int pr = pivrow_s;
        if (pr != p && tid < NFERM){
            float tmp = As[p][tid]; As[p][tid] = As[pr][tid]; As[pr][tid] = tmp;
        }
        __syncthreads();
        if (tid == 0){
            float piv = As[p][p];
            la += logf(fabsf(piv));
            if (piv < 0.f) sg = -sg;
            if (pr != p)   sg = -sg;
            pivinv_s = 1.0f/piv;
        }
        __syncthreads();
        float pinv = pivinv_s;
        int nr = NFERM - 1 - p;
        for (int e = tid; e < nr*NFERM; e += 256){
            int r = p + 1 + (e >> 6), c = e & 63;
            if (c > p) As[r][c] -= As[r][p]*pinv*As[p][c];
        }
        __syncthreads();
    }
    if (tid == 0){ logabs[bk] = la; sgn[bk] = (float)sg; }
}

// ---------------- k4: stable complex logsumexp over K determinants ----------------
__global__ void k4_lse(const float* __restrict__ logabs, const float* __restrict__ sgn,
                       const float* __restrict__ lc, float* __restrict__ out){
    int b = blockIdx.x*blockDim.x + threadIdx.x;
    if (b >= B_SZ) return;
    float re[KENS], sv[KENS];
    float m = -INFINITY;
    for (int k = 0; k < KENS; ++k){
        re[k] = logabs[b*KENS + k] + lc[k];
        sv[k] = sgn[b*KENS + k];
        if (re[k] > m) m = re[k];
    }
    float ssum = 0.f;
    for (int k = 0; k < KENS; ++k) ssum += sv[k]*expf(re[k] - m);   // sign = e^{i*pi}
    out[2*b + 0] = logf(fabsf(ssum)) + m;
    out[2*b + 1] = (ssum < 0.f) ? PI_F : 0.f;
}

extern "C" void kernel_launch(void* const* d_in, const int* in_sizes, int n_in,
                              void* d_out, int out_size, void* d_ws, size_t ws_size,
                              hipStream_t stream) {
    const int*   n_dev   = (const int*)  d_in[0];
    const float* W_lift  = (const float*)d_in[1];
    const float* b_lift  = (const float*)d_in[2];
    const float* Wsp_re  = (const float*)d_in[3];
    const float* Wsp_im  = (const float*)d_in[4];
    const float* W_pt    = (const float*)d_in[5];
    const float* b_pt    = (const float*)d_in[6];
    const float* W_proj  = (const float*)d_in[7];
    const float* b_proj  = (const float*)d_in[8];
    const float* log_co  = (const float*)d_in[9];
    float* out = (float*)d_out;   // complex64 interleaved (re, im) per sample

    char* ws = (char*)d_ws;
    _Float16* T_ws   = (_Float16*)(ws + 0);          // 512*128*2      = 131072 B
    float* Wc_ws     = (float*)(ws + 131072);        // 4*64*64*2*4    = 131072 B
    float* bterm_ws  = (float*)(ws + 262144);        // 4*64*2*4       = 2048 B
    float* wp_ws     = (float*)(ws + 264192);        // 1024 B
    float* bp2_ws    = (float*)(ws + 265216);        // 1024 B
    float* nf_ws     = (float*)(ws + 266240);        // 1024*64*2*4    = 524288 B
    int*   R_ws      = (int*)  (ws + 790528);        // 1024*64*4      = 262144 B
    float* feat_ws   = (float*)(ws + 1052672);       // 1024*4*64*4    = 1048576 B
    float* logabs_ws = (float*)(ws + 2101248);       // 4096*4         = 16384 B
    float* sgn_ws    = (float*)(ws + 2117632);       // 4096*4         = 16384 B

    k0_modes_gather<<<B_SZ, 64, 0, stream>>>(n_dev, nf_ws, R_ws);
    k1_weights<<<64, 256, 0, stream>>>(W_lift, b_lift, Wsp_re, Wsp_im, W_pt, b_pt,
                                       Wc_ws, bterm_ws, wp_ws, bp2_ws);
    kT_basis<<<256, 256, 0, stream>>>(T_ws);
    k2_fno_feat<<<B_SZ*KENS, 256, 0, stream>>>(T_ws, Wc_ws, bterm_ws, wp_ws, bp2_ws,
                                               nf_ws, n_dev, feat_ws);
    k3_det<<<B_SZ*KENS, 256, 0, stream>>>(feat_ws, R_ws, W_proj, b_proj, logabs_ws, sgn_ws);
    k4_lse<<<(B_SZ + 255)/256, 256, 0, stream>>>(logabs_ws, sgn_ws, log_co, out);
}
